// MetalQuantizedMoE_11862699671917
// MI455X (gfx1250) — matrix operand order repaired and checked by reading, not compiled, not run on hardware
//
#include <hip/hip_runtime.h>
#include <hip/hip_fp16.h>

#define T_TOK   2048
#define K_DIM   2048
#define I_DIM   1024
#define N_EXP   8
#define NROUTED (T_TOK * 2)        // T * topk = 4096
#define NACT    (NROUTED + T_TOK)  // + shared expert rows = 6144
#define MT      4                  // M tiles per block (64 rows)
#define MROWS   (MT * 16)
#define KCH     256                // K chunk staged in LDS
#define N2      (2 * I_DIM)

// workspace byte offsets
#define OFF_CNTS 0
#define OFF_OFFS 64
#define OFF_TOK  128
#define OFF_PRB  (OFF_TOK + NACT * 4)
#define OFF_XH   49408ull                               // T*K f16 = 8 MB
#define OFF_ACT  (OFF_XH + (size_t)T_TOK * K_DIM * 2)   // NACT*I f16 = 12 MB
#define OFF_WGU  (OFF_ACT + (size_t)NACT * I_DIM * 2)   // 9 experts swizzled gate_up f16
#define GU_BLK   ((N2 / 16) * (K_DIM / 32))             // 8192 tiles/expert
#define DN_BLK   ((K_DIM / 16) * (I_DIM / 32))          // 4096 tiles/expert
#define OFF_WDN  (OFF_WGU + (size_t)9 * GU_BLK * 1024)

typedef __attribute__((ext_vector_type(16))) _Float16 v16h;
typedef __attribute__((ext_vector_type(8)))  float    v8f;

union HFrag {
    v16h     v;
    uint4    q[2];
    unsigned u[8];
};

// FP4 E2M1 nibble -> f16 bits (sign + {0,.5,1,1.5,2,3,4,6})
__device__ __forceinline__ unsigned fp4_to_h16(unsigned nib) {
    unsigned m    = nib & 7u;
    unsigned frac = ((m & 1u) & (unsigned)(m > 2u)) << 9;
    unsigned bits = ((14u + (m >> 1)) << 10) | frac;
    bits = m ? bits : 0u;
    return bits | ((nib & 8u) << 12);
}

__device__ __forceinline__ v16h decode16(unsigned w0, unsigned w1, __half2 s2) {
    HFrag f;
#pragma unroll
    for (int j = 0; j < 4; ++j) {
        unsigned n0 = (w0 >> (8 * j)) & 0xFu, n1 = (w0 >> (8 * j + 4)) & 0xFu;
        unsigned p  = fp4_to_h16(n0) | (fp4_to_h16(n1) << 16);
        __half2 hv  = __hmul2(*reinterpret_cast<const __half2*>(&p), s2);
        f.u[j] = *reinterpret_cast<const unsigned*>(&hv);
    }
#pragma unroll
    for (int j = 0; j < 4; ++j) {
        unsigned n0 = (w1 >> (8 * j)) & 0xFu, n1 = (w1 >> (8 * j + 4)) & 0xFu;
        unsigned p  = fp4_to_h16(n0) | (fp4_to_h16(n1) << 16);
        __half2 hv  = __hmul2(*reinterpret_cast<const __half2*>(&p), s2);
        f.u[4 + j] = *reinterpret_cast<const unsigned*>(&hv);
    }
    return f.v;
}

// ---------------- routing ----------------
__global__ __launch_bounds__(256) void k_route(const int* __restrict__ eids,
                                               const float* __restrict__ probs,
                                               int* cnts, int* offs, int* tok, float* prb) {
    __shared__ int cnt[N_EXP], cur[N_EXP];
    int tid = threadIdx.x;
    if (tid < N_EXP) cnt[tid] = 0;
    __syncthreads();
    for (int i = tid; i < NROUTED; i += 256) atomicAdd(&cnt[eids[i]], 1);
    __syncthreads();
    if (tid == 0) {
        int o = 0;
        for (int e = 0; e < N_EXP; ++e) { offs[e] = o; cur[e] = o; cnts[e] = cnt[e]; o += cnt[e]; }
        offs[N_EXP] = NROUTED;
        cnts[N_EXP] = T_TOK;
    }
    __syncthreads();
    for (int i = tid; i < NROUTED; i += 256) {
        int e = eids[i];
        int p = atomicAdd(&cur[e], 1);
        tok[p] = i >> 1;
        prb[p] = probs[i];
    }
    for (int t = tid; t < T_TOK; t += 256) { tok[NROUTED + t] = t; prb[NROUTED + t] = 1.0f; }
}

__global__ __launch_bounds__(256) void k_zero(float4* __restrict__ out, int n4) {
    int i = blockIdx.x * 256 + threadIdx.x;
    if (i < n4) out[i] = make_float4(0.f, 0.f, 0.f, 0.f);
}

// ---------------- x: f32 -> f16 row-major ----------------
__global__ __launch_bounds__(256) void k_xhalf(const float* __restrict__ x,
                                               _Float16* __restrict__ xh) {
    int i = blockIdx.x * 256 + threadIdx.x;          // one float4 each
    float4 v = ((const float4*)x)[i];
    __half2* p = (__half2*)(xh + (size_t)i * 4);
    p[0] = __floats2half2_rn(v.x, v.y);
    p[1] = __floats2half2_rn(v.z, v.w);
}

// ---------------- dequant FP4 -> f16 in swizzled B-fragment tiles ----------------
// Tile (nt, ks32) of W[Kd][Nd]: 1KB block; lane l holds cols nt*16+(l&15),
// K = ks32*32 + (l>>4)*16 .. +15, stored as 16 contiguous halfs at lane*32B.
__global__ __launch_bounds__(256) void k_dequant(const int* __restrict__ pk,
                                                 const float* __restrict__ sc,
                                                 const int* __restrict__ spk,
                                                 const float* __restrict__ ssc,
                                                 _Float16* __restrict__ outw,
                                                 int Kd, int Nd) {
    int z = blockIdx.z;
    const int*   pz = (z < N_EXP) ? pk + (size_t)z * (Kd / 8)  * Nd : spk;
    const float* sz = (z < N_EXP) ? sc + (size_t)z * (Kd / 64) * Nd : ssc;
    int c    = blockIdx.x * 256 + threadIdx.x;
    int lane = c & 31, blk = c >> 5;
    int ks32 = blk % (Kd / 32), nt = blk / (Kd / 32);
    int nn = lane & 15, kh = lane >> 4;
    int n  = nt * 16 + nn;
    int kk = ks32 * 32 + kh * 16;
    int r8 = kk >> 3;
    unsigned w0 = (unsigned)pz[(size_t)r8       * Nd + n];
    unsigned w1 = (unsigned)pz[(size_t)(r8 + 1) * Nd + n];
    float    s  = sz[(size_t)(kk >> 6) * Nd + n];
    HFrag f;
    f.v = decode16(w0, w1, __float2half2_rn(s));
    size_t nblk = (size_t)(Nd / 16) * (Kd / 32);
    uint4* dst = (uint4*)(outw + ((size_t)z * nblk + blk) * 512 + lane * 16);
    dst[0] = f.q[0];
    dst[1] = f.q[1];
}

// ---------------- GEMM1: act = silu(x Wg) * (x Wu) ----------------
__global__ __launch_bounds__(256) void k_gemm1(
    const _Float16* __restrict__ xh, const _Float16* __restrict__ wgu,
    const int* __restrict__ cnts, const int* __restrict__ offs,
    const int* __restrict__ tok, _Float16* __restrict__ act) {
    __shared__ __align__(16) _Float16 xs[MROWS * KCH];   // 32 KB
    __shared__ int stok[MROWS];

    const int z = blockIdx.z, mt = blockIdx.y, cg = blockIdx.x;
    const int count = cnts[z];
    if (mt * MROWS >= count) return;                     // uniform
    const int lbase = offs[z];

    const int tid = threadIdx.x;
    if (tid < MROWS) {
        int r = mt * MROWS + tid;
        stok[tid] = (r < count) ? tok[lbase + r] : -1;
    }

    const int lane = tid & 31, w = tid >> 5;
    const int n0 = cg * 128 + w * 16;                    // gate column base within I
    const int nn = lane & 15, kh = lane >> 4;

    // swizzled weight base for this wave's column tile (gate) and (up = + I/16 tiles)
    const _Float16* bg0 = wgu + ((size_t)z * GU_BLK + (size_t)(cg * 8 + w) * (K_DIM / 32)) * 512
                              + lane * 16;
    const _Float16* bu0 = bg0 + (size_t)(I_DIM / 16) * (K_DIM / 32) * 512;

    v8f accg[MT], accu[MT];
#pragma unroll
    for (int t = 0; t < MT; ++t) { accg[t] = (v8f){}; accu[t] = (v8f){}; }

    for (int kc = 0; kc < K_DIM; kc += KCH) {
        __syncthreads();
        // stage MROWS x KCH halfs (pure b128 copy): 32 uint4 per row
        for (int idx = tid; idx < MROWS * (KCH / 8); idx += 256) {
            int r = idx >> 5, cc = idx & 31;
            int t = stok[r];
            uint4 v = make_uint4(0, 0, 0, 0);
            if (t >= 0) v = ((const uint4*)(xh + (size_t)t * K_DIM + kc))[cc];
            ((uint4*)(xs + r * KCH))[cc] = v;
        }
        __syncthreads();

        for (int ks = 0; ks < KCH; ks += 32) {
            const int kidx = (kc + ks) >> 5;
            HFrag bg, bu;
            const uint4* pg = (const uint4*)(bg0 + (size_t)kidx * 512);
            const uint4* pu = (const uint4*)(bu0 + (size_t)kidx * 512);
            bg.q[0] = pg[0]; bg.q[1] = pg[1];
            bu.q[0] = pu[0]; bu.q[1] = pu[1];

            // load ALL A fragments first so the DS clause can drain incrementally
            HFrag a[MT];
#pragma unroll
            for (int t = 0; t < MT; ++t) {
                const uint4* ap = (const uint4*)(xs + (t * 16 + nn) * KCH + ks + kh * 16);
                a[t].q[0] = ap[0];
                a[t].q[1] = ap[1];
            }
#pragma unroll
            for (int t = 0; t < MT; ++t) {
                accg[t] = __builtin_amdgcn_wmma_f32_16x16x32_f16(false, a[t].v, false, bg.v,
                                                                 (short)0, accg[t], false, false);
                accu[t] = __builtin_amdgcn_wmma_f32_16x16x32_f16(false, a[t].v, false, bu.v,
                                                                 (short)0, accu[t], false, false);
            }
        }
    }

    const int Mb = kh << 3;
#pragma unroll
    for (int t = 0; t < MT; ++t) {
#pragma unroll
        for (int r = 0; r < 8; ++r) {
            int M = t * 16 + Mb + r, row = mt * MROWS + M;
            float g = accg[t][r], u = accu[t][r];
            float av = g * u * __builtin_amdgcn_rcpf(1.0f + __expf(-g));
            if (row < count)
                act[(size_t)(lbase + row) * I_DIM + (n0 + nn)] = (_Float16)av;
        }
    }
}

// ---------------- GEMM2: out += prob * (act Wd) ----------------
__global__ __launch_bounds__(256) void k_gemm2(
    const _Float16* __restrict__ act, const _Float16* __restrict__ wdn,
    const int* __restrict__ cnts, const int* __restrict__ offs,
    const int* __restrict__ tok, const float* __restrict__ prb,
    float* __restrict__ out) {
    __shared__ __align__(16) _Float16 as[MROWS * KCH];   // 32 KB
    __shared__ int stok[MROWS];
    __shared__ float sp[MROWS];

    const int z = blockIdx.z, mt = blockIdx.y, cg = blockIdx.x;
    const int count = cnts[z];
    if (mt * MROWS >= count) return;
    const int lbase = offs[z];

    const int tid = threadIdx.x;
    if (tid < MROWS) {
        int r = mt * MROWS + tid;
        stok[tid] = (r < count) ? tok[lbase + r] : -1;
        sp[tid]   = (r < count) ? prb[lbase + r] : 0.f;
    }

    const int lane = tid & 31, w = tid >> 5;
    const int n0 = cg * 128 + w * 16, nn = lane & 15, kh = lane >> 4;

    const _Float16* bd0 = wdn + ((size_t)z * DN_BLK + (size_t)(cg * 8 + w) * (I_DIM / 32)) * 512
                              + lane * 16;

    v8f acc[MT];
#pragma unroll
    for (int t = 0; t < MT; ++t) acc[t] = (v8f){};

    for (int ic = 0; ic < I_DIM; ic += KCH) {
        __syncthreads();
        for (int idx = tid; idx < MROWS * (KCH / 8); idx += 256) {
            int r = idx >> 5, cc = idx & 31;
            uint4 v = make_uint4(0, 0, 0, 0);
            if (mt * MROWS + r < count)
                v = ((const uint4*)(act + (size_t)(lbase + mt * MROWS + r) * I_DIM + ic))[cc];
            ((uint4*)(as + r * KCH))[cc] = v;
        }
        __syncthreads();

        for (int ks = 0; ks < KCH; ks += 32) {
            const int kidx = (ic + ks) >> 5;
            HFrag b;
            const uint4* pb = (const uint4*)(bd0 + (size_t)kidx * 512);
            b.q[0] = pb[0]; b.q[1] = pb[1];

            HFrag a[MT];
#pragma unroll
            for (int t = 0; t < MT; ++t) {
                const uint4* ap = (const uint4*)(as + (t * 16 + nn) * KCH + ks + kh * 16);
                a[t].q[0] = ap[0];
                a[t].q[1] = ap[1];
            }
#pragma unroll
            for (int t = 0; t < MT; ++t) {
                acc[t] = __builtin_amdgcn_wmma_f32_16x16x32_f16(false, a[t].v, false, b.v,
                                                                (short)0, acc[t], false, false);
            }
        }
    }

    const int Mb = kh << 3;
#pragma unroll
    for (int t = 0; t < MT; ++t) {
#pragma unroll
        for (int r = 0; r < 8; ++r) {
            int M = t * 16 + Mb + r, row = mt * MROWS + M;
            if (row < count) {
                int tk = stok[M];
                unsafeAtomicAdd(out + (size_t)tk * K_DIM + (n0 + nn), acc[t][r] * sp[M]);
            }
        }
    }
}

extern "C" void kernel_launch(void* const* d_in, const int* in_sizes, int n_in,
                              void* d_out, int out_size, void* d_ws, size_t ws_size,
                              hipStream_t stream) {
    const float* x    = (const float*)d_in[0];
    const int*   gup  = (const int*)d_in[1];
    const float* gus  = (const float*)d_in[2];
    const int*   dwp  = (const int*)d_in[3];
    const float* dws  = (const float*)d_in[4];
    const int*   sgup = (const int*)d_in[5];
    const float* sgus = (const float*)d_in[6];
    const int*   sdwp = (const int*)d_in[7];
    const float* sdws = (const float*)d_in[8];
    const int*   eids = (const int*)d_in[9];
    const float* epr  = (const float*)d_in[10];
    float* out = (float*)d_out;

    char* ws = (char*)d_ws;
    int*      cnts = (int*)(ws + OFF_CNTS);
    int*      offs = (int*)(ws + OFF_OFFS);
    int*      tok  = (int*)(ws + OFF_TOK);
    float*    prb  = (float*)(ws + OFF_PRB);
    _Float16* xh   = (_Float16*)(ws + OFF_XH);
    _Float16* act  = (_Float16*)(ws + OFF_ACT);
    _Float16* wgu  = (_Float16*)(ws + OFF_WGU);
    _Float16* wdn  = (_Float16*)(ws + OFF_WDN);

    k_route<<<1, 256, 0, stream>>>(eids, epr, cnts, offs, tok, prb);
    k_xhalf<<<T_TOK * K_DIM / 4 / 256, 256, 0, stream>>>(x, xh);
    dim3 gq1(GU_BLK * 32 / 256, 1, N_EXP + 1);          // (1024, 1, 9)
    k_dequant<<<gq1, 256, 0, stream>>>(gup, gus, sgup, sgus, wgu, K_DIM, N2);
    dim3 gq2(DN_BLK * 32 / 256, 1, N_EXP + 1);          // (512, 1, 9)
    k_dequant<<<gq2, 256, 0, stream>>>(dwp, dws, sdwp, sdws, wdn, I_DIM, K_DIM);
    k_zero<<<(T_TOK * K_DIM / 4 + 255) / 256, 256, 0, stream>>>((float4*)out,
                                                                T_TOK * K_DIM / 4);
    dim3 g1(I_DIM / 128, NROUTED / MROWS, N_EXP + 1);   // (8, 64, 9)
    k_gemm1<<<g1, 256, 0, stream>>>(xh, wgu, cnts, offs, tok, act);
    dim3 g2(K_DIM / 128, NROUTED / MROWS, N_EXP + 1);   // (16, 64, 9)
    k_gemm2<<<g2, 256, 0, stream>>>(act, wdn, cnts, offs, tok, prb, out);
}
